// MoELayer_53919019434507
// MI455X (gfx1250) — matrix operand order repaired and checked by reading, not compiled, not run on hardware
//
#include <hip/hip_runtime.h>
#include <hip/hip_bf16.h>
#include <math.h>

// ---------------- problem constants ----------------
#define T_TOKENS 16384   // 4*4096
#define D_MODEL  512
#define HIDDEN   2048
#define NEXP     8
#define CAP      16384   // max tokens per expert (top-2 distinct => <= T)
#define MTILE    32      // tokens per workgroup tile (B-fragment reuse = 32 rows)
#define MAX_TILES ((2*T_TOKENS)/MTILE + NEXP)   // 1032

typedef __attribute__((ext_vector_type(16))) __bf16 v16bf;
typedef __attribute__((ext_vector_type(8)))  float  v8f;

// ---------------- WMMA fragment loaders (bf16, 16x16x32) ----------------
// A matrix 16x32 (MxK), 16-bit layout per CDNA5 ISA:
//  lanes 0-15 : M=lane,    K = kc+{0..7} and kc+{16..23}
//  lanes 16-31: M=lane-16, K = kc+{8..15} and kc+{24..31}
__device__ __forceinline__ v16bf loadA_frag(const __bf16* __restrict__ buf, int ld,
                                            int kc, int lane) {
  int m    = lane & 15;
  int half = lane >> 4;
  const __bf16* p0 = buf + (size_t)m * ld + kc + half * 8;
  union { v16bf v; uint4 q[2]; } u;
  u.q[0] = *(const uint4*)(p0);
  u.q[1] = *(const uint4*)(p0 + 16);
  return u.v;
}

// B matrix 32x16 (KxN), 16-bit layout matching A:
//  lanes 0-15 : N=lane,    K = kc+{0..7} and kc+{16..23}
//  lanes 16-31: N=lane-16, K = kc+{8..15} and kc+{24..31}
// Wt is stored [N][K] row-major so each 8 K values are contiguous.
__device__ __forceinline__ v16bf loadB_frag(const __bf16* __restrict__ Wt, int K,
                                            int n0, int kc, int lane) {
  int n    = n0 + (lane & 15);
  int half = lane >> 4;
  const __bf16* p = Wt + (size_t)n * K + kc + half * 8;
  union { v16bf v; uint4 q[2]; } u;
  u.q[0] = *(const uint4*)(p);
  u.q[1] = *(const uint4*)(p + 16);
  return u.v;
}

__device__ __forceinline__ v8f wmma_bf16(v16bf a, v16bf b, v8f c) {
  return __builtin_amdgcn_wmma_f32_16x16x32_bf16(false, a, false, b, (short)0, c,
                                                 false, false);
}

// ---------------- conversion kernels ----------------
__global__ void k_cvt_x(const float* __restrict__ x, __bf16* __restrict__ xb) {
  size_t i = (size_t)blockIdx.x * blockDim.x + threadIdx.x;
  if (i < (size_t)T_TOKENS * D_MODEL) xb[i] = (__bf16)x[i];
}

// W1[e][d][h] -> W1t[e][h][d] (bf16)
__global__ void k_cvt_w1(const float* __restrict__ W1, __bf16* __restrict__ W1t) {
  size_t i = (size_t)blockIdx.x * blockDim.x + threadIdx.x;
  if (i >= (size_t)NEXP * HIDDEN * D_MODEL) return;
  int d = (int)(i & (D_MODEL - 1));
  size_t r = i >> 9;                       // /512
  int h = (int)(r & (HIDDEN - 1));
  int e = (int)(r >> 11);                  // /2048
  W1t[i] = (__bf16)W1[((size_t)e * D_MODEL + d) * HIDDEN + h];
}

// W2[e][h][d] -> W2t[e][d][h] (bf16)
__global__ void k_cvt_w2(const float* __restrict__ W2, __bf16* __restrict__ W2t) {
  size_t i = (size_t)blockIdx.x * blockDim.x + threadIdx.x;
  if (i >= (size_t)NEXP * D_MODEL * HIDDEN) return;
  int h = (int)(i & (HIDDEN - 1));
  size_t r = i >> 11;                      // /2048
  int d = (int)(r & (D_MODEL - 1));
  int e = (int)(r >> 9);                   // /512
  W2t[i] = (__bf16)W2[((size_t)e * HIDDEN + h) * D_MODEL + d];
}

// ---------------- router: logits, top-2 softmax, gather lists ----------------
__global__ void k_router(const float* __restrict__ x, const float* __restrict__ Wg,
                         float* __restrict__ logits_out, int* __restrict__ counts,
                         int* __restrict__ tokList, float* __restrict__ wList) {
  __shared__ float wgs[NEXP * D_MODEL];    // 16 KB
  int tid = threadIdx.x;
  for (int i = tid; i < NEXP * D_MODEL; i += 256) wgs[i] = Wg[i];
  __syncthreads();

  int wave = tid >> 5, lane = tid & 31;
  int t = blockIdx.x * 8 + wave;           // one wave per token
  if (t >= T_TOKENS) return;

  float acc[NEXP];
#pragma unroll
  for (int e = 0; e < NEXP; ++e) acc[e] = 0.0f;
  const float* xr = x + (size_t)t * D_MODEL;
  for (int j = lane; j < D_MODEL; j += 32) {
    float xv = xr[j];
#pragma unroll
    for (int e = 0; e < NEXP; ++e) acc[e] += xv * wgs[e * D_MODEL + j];
  }
#pragma unroll
  for (int e = 0; e < NEXP; ++e) {
#pragma unroll
    for (int off = 16; off > 0; off >>= 1)
      acc[e] += __shfl_down(acc[e], off, 32);
  }
  if (lane == 0) {
    float v0 = -INFINITY; int e0 = 0;
#pragma unroll
    for (int e = 0; e < NEXP; ++e) if (acc[e] > v0) { v0 = acc[e]; e0 = e; }
    float v1 = -INFINITY; int e1 = 0;
#pragma unroll
    for (int e = 0; e < NEXP; ++e)
      if (e != e0 && acc[e] > v1) { v1 = acc[e]; e1 = e; }
#pragma unroll
    for (int e = 0; e < NEXP; ++e) logits_out[(size_t)t * NEXP + e] = acc[e];
    float ex = expf(v1 - v0);               // v1 <= v0 : stable
    float w0 = 1.0f / (1.0f + ex);
    float w1 = ex / (1.0f + ex);
    int p0 = atomicAdd(&counts[e0], 1);
    tokList[(size_t)e0 * CAP + p0] = t; wList[(size_t)e0 * CAP + p0] = w0;
    int p1 = atomicAdd(&counts[e1], 1);
    tokList[(size_t)e1 * CAP + p1] = t; wList[(size_t)e1 * CAP + p1] = w1;
  }
}

// ---------------- tile prefix sums ----------------
__global__ void k_meta(const int* __restrict__ counts, int* __restrict__ tileStart) {
  if (threadIdx.x == 0 && blockIdx.x == 0) {
    int tot = 0;
    for (int e = 0; e < NEXP; ++e) {
      tileStart[e] = tot;
      tot += (counts[e] + MTILE - 1) / MTILE;
    }
    tileStart[NEXP] = tot;
  }
}

// ---------------- fused expert MLP (grouped GEMM, 32-token tiles) ----------------
// Block = 256 threads = 8 waves. One block = one 32-token tile of one expert.
// Phase A: h[32,2048] = gelu(x_tile[32,512] @ W1t[e]^T) -> LDS (bf16)
// Phase B: y[32,512]  = h @ W2t[e]^T, scaled by routing weight, atomicAdd to out.
// Each B fragment (weights) is reused by 2 M-tiles -> 32 MACs/weight element.
__global__ void k_moe_mlp(const __bf16* __restrict__ xb,
                          const __bf16* __restrict__ W1t,
                          const __bf16* __restrict__ W2t,
                          const int* __restrict__ tokList,
                          const float* __restrict__ wList,
                          const int* __restrict__ counts,
                          const int* __restrict__ tileStart,
                          float* __restrict__ out) {
  extern __shared__ __align__(16) char smem[];
  __bf16* x_lds = (__bf16*)smem;                                 // 32*512*2  =  32 KB
  __bf16* h_lds = (__bf16*)(smem + MTILE * D_MODEL * 2);         // 32*2048*2 = 128 KB
  int*    stok  = (int*)(smem + MTILE * D_MODEL * 2 + MTILE * HIDDEN * 2);
  float*  swt   = (float*)(stok + MTILE);

  int b = blockIdx.x;
  if (b >= tileStart[NEXP]) return;
  int e = 0;
  while (e < NEXP - 1 && b >= tileStart[e + 1]) ++e;
  int tile = b - tileStart[e];
  int cnt  = counts[e];

  int tid = threadIdx.x;
  if (tid < MTILE) {
    int idx = tile * MTILE + tid;
    const int*   tl = tokList + (size_t)e * CAP;
    const float* wl = wList  + (size_t)e * CAP;
    if (idx < cnt) { stok[tid] = tl[idx]; swt[tid] = wl[idx]; }
    else           { stok[tid] = tl[0];   swt[tid] = 0.0f;   }
  }
  __syncthreads();

  // stage x tile into LDS (32 rows x 512 bf16 = 32 rows x 256 dwords)
  {
    const unsigned* xb32 = (const unsigned*)xb;
    unsigned* xl32 = (unsigned*)x_lds;
    for (int i = tid; i < MTILE * (D_MODEL / 2); i += 256) {
      int row = i >> 8, col = i & 255;
      xl32[i] = xb32[(size_t)stok[row] * (D_MODEL / 2) + col];
    }
  }
  __syncthreads();

  int wave = tid >> 5, lane = tid & 31;
  int m0 = (lane >> 4) * 8;   // C/D row base for this lane (within a 16-row M-tile)
  int nc = lane & 15;         // C/D column offset

  // ---- Phase A: h = gelu(x @ W1e), wave owns 256 hidden columns ----
  const __bf16* W1e = W1t + (size_t)e * HIDDEN * D_MODEL;
  for (int g = 0; g < 4; ++g) {
    int nb = wave * 256 + g * 64;
    v8f acc[2][4];
#pragma unroll
    for (int mh = 0; mh < 2; ++mh)
#pragma unroll
      for (int j = 0; j < 4; ++j) acc[mh][j] = (v8f){0,0,0,0,0,0,0,0};
    for (int kc = 0; kc < D_MODEL; kc += 32) {
      v16bf a0 = loadA_frag(x_lds,                 D_MODEL, kc, lane);
      v16bf a1 = loadA_frag(x_lds + 16 * D_MODEL,  D_MODEL, kc, lane);
#pragma unroll
      for (int j = 0; j < 4; ++j) {
        v16bf bb = loadB_frag(W1e, D_MODEL, nb + 16 * j, kc, lane);
        acc[0][j] = wmma_bf16(a0, bb, acc[0][j]);
        acc[1][j] = wmma_bf16(a1, bb, acc[1][j]);
      }
    }
#pragma unroll
    for (int mh = 0; mh < 2; ++mh) {
#pragma unroll
      for (int j = 0; j < 4; ++j) {
        int n = nb + 16 * j + nc;
#pragma unroll
        for (int r = 0; r < 8; ++r) {
          float v  = acc[mh][j][r];
          float gl = 0.5f * v * (1.0f + erff(v * 0.70710678118654752440f)); // exact GELU
          h_lds[(size_t)(mh * 16 + m0 + r) * HIDDEN + n] = (__bf16)gl;
        }
      }
    }
  }
  __syncthreads();

  // ---- Phase B: y = h @ W2e, wave owns 64 output columns ----
  const __bf16* W2e = W2t + (size_t)e * D_MODEL * HIDDEN;
  int nb2 = wave * 64;
  v8f acc[2][4];
#pragma unroll
  for (int mh = 0; mh < 2; ++mh)
#pragma unroll
    for (int j = 0; j < 4; ++j) acc[mh][j] = (v8f){0,0,0,0,0,0,0,0};
  for (int kc = 0; kc < HIDDEN; kc += 32) {
    v16bf a0 = loadA_frag(h_lds,                HIDDEN, kc, lane);
    v16bf a1 = loadA_frag(h_lds + 16 * HIDDEN,  HIDDEN, kc, lane);
#pragma unroll
    for (int j = 0; j < 4; ++j) {
      v16bf bb = loadB_frag(W2e, HIDDEN, nb2 + 16 * j, kc, lane);
      acc[0][j] = wmma_bf16(a0, bb, acc[0][j]);
      acc[1][j] = wmma_bf16(a1, bb, acc[1][j]);
    }
  }
#pragma unroll
  for (int mh = 0; mh < 2; ++mh) {
#pragma unroll
    for (int j = 0; j < 4; ++j) {
      int n = nb2 + 16 * j + nc;
#pragma unroll
      for (int r = 0; r < 8; ++r) {
        int m = mh * 16 + m0 + r;
        atomicAdd(out + (size_t)stok[m] * D_MODEL + n, acc[mh][j][r] * swt[m]);
      }
    }
  }
}

// ---------------- host-side orchestration ----------------
extern "C" void kernel_launch(void* const* d_in, const int* in_sizes, int n_in,
                              void* d_out, int out_size, void* d_ws, size_t ws_size,
                              hipStream_t stream) {
  (void)in_sizes; (void)n_in; (void)out_size; (void)ws_size;
  const float* x  = (const float*)d_in[0];
  const float* Wg = (const float*)d_in[1];
  const float* W1 = (const float*)d_in[2];
  const float* W2 = (const float*)d_in[3];
  float* out    = (float*)d_out;                                   // [T, 512]
  float* logits = (float*)d_out + (size_t)T_TOKENS * D_MODEL;      // [T, 8]

  // workspace layout (16B aligned slabs)
  char* ws = (char*)d_ws;
  __bf16* xb   = (__bf16*)ws;                                      // 16 MB
  __bf16* W1t  = (__bf16*)(ws + (size_t)16 * 1024 * 1024);         // 16 MB
  __bf16* W2t  = (__bf16*)(ws + (size_t)32 * 1024 * 1024);         // 16 MB
  int*    tokL = (int*)   (ws + (size_t)48 * 1024 * 1024);         // 512 KB
  float*  wL   = (float*) (ws + (size_t)48 * 1024 * 1024 + 524288);// 512 KB
  int*    cnts = (int*)   (ws + (size_t)48 * 1024 * 1024 + 1048576);  // 8 ints
  int*    tst  = cnts + NEXP;                                      // 9 ints

  hipMemsetAsync(out, 0, (size_t)T_TOKENS * D_MODEL * sizeof(float), stream);
  hipMemsetAsync(cnts, 0, (NEXP + NEXP + 1) * sizeof(int), stream);

  const size_t NW = (size_t)NEXP * D_MODEL * HIDDEN;   // 8,388,608
  k_cvt_x <<<(unsigned)((T_TOKENS * (size_t)D_MODEL + 255) / 256), 256, 0, stream>>>(x, xb);
  k_cvt_w1<<<(unsigned)((NW + 255) / 256), 256, 0, stream>>>(W1, W1t);
  k_cvt_w2<<<(unsigned)((NW + 255) / 256), 256, 0, stream>>>(W2, W2t);

  k_router<<<T_TOKENS / 8, 256, 0, stream>>>(x, Wg, logits, cnts, tokL, wL);
  k_meta<<<1, 32, 0, stream>>>(cnts, tst);

  const size_t smem_bytes = (size_t)MTILE * D_MODEL * 2   // x_lds  (32 KB)
                          + (size_t)MTILE * HIDDEN * 2    // h_lds (128 KB)
                          + MTILE * sizeof(int) + MTILE * sizeof(float);
  k_moe_mlp<<<MAX_TILES, 256, smem_bytes, stream>>>(xb, W1t, W2t, tokL, wL,
                                                    cnts, tst, out);
}
